// Memory_Transformer_12043088298476
// MI455X (gfx1250) — matrix operand order repaired and checked by reading, not compiled, not run on hardware
//
#include <hip/hip_runtime.h>

// ---------------------------------------------------------------------------
// Causal single-head attention, T=8192, d=1024, fp32 in/out.
// Compute-bound => bf16 WMMA (v_wmma_f32_16x16x32_bf16) with f32 accum.
// Flash-attention single pass; K/V converted to bf16 (16MB each -> L2-resident).
// Block = 256 threads (8 waves, wave32). Block owns 16 Q rows.
//   - S = Q K^T : waves split d (128-wide slices), partial S reduced via LDS.
//   - softmax   : distributed — wave w owns rows {2w, 2w+1}; lanes span the
//                 32 KV columns, so row reductions are 5-step __shfl_xor.
//   - O = P V   : waves split V columns (128 each), O in registers (64 VGPR).
// Both GEMM phases preload ALL 8 B-fragments before the WMMA chain; an empty
// asm register-tie pins the K fragments in distinct VGPRs so the scheduler
// cannot re-serialize the loads (1 exposed L2 latency per tile, not 8).
// Workspace: Q_bf16 (16MB) + K_bf16 (16MB) + V^T_bf16 (16MB) = 48MB.
// ---------------------------------------------------------------------------

typedef __bf16 bf16_t;
typedef __attribute__((ext_vector_type(16))) __bf16 v16bf;
typedef __attribute__((ext_vector_type(8)))  __bf16 v8bf;
typedef __attribute__((ext_vector_type(4)))  __bf16 v4bf;
typedef __attribute__((ext_vector_type(8)))  float  v8f;

union ABFrag { v16bf v; v8bf h[2]; };

__device__ __forceinline__ bf16_t f2bf(float x) {
  unsigned u = __builtin_bit_cast(unsigned, x);
  u += 0x7FFFu + ((u >> 16) & 1u);            // round-to-nearest-even
  unsigned short hv = (unsigned short)(u >> 16);
  return __builtin_bit_cast(bf16_t, hv);
}

constexpr int Tt = 8192;   // sequence length
constexpr int Dd = 1024;   // head dim (= value dim)

// ---- fp32 -> bf16, row-major (for Q and K) --------------------------------
__global__ void cvt_rm_kernel(const float* __restrict__ in, bf16_t* __restrict__ out) {
  size_t i = ((size_t)blockIdx.x * 256u + threadIdx.x) * 4u;
  const float4 f = *(const float4*)(in + i);
  v4bf o4 = { f2bf(f.x), f2bf(f.y), f2bf(f.z), f2bf(f.w) };
  *(v4bf*)(out + i) = o4;
}

// ---- fp32 V[s][c] -> bf16 V^T[c][s] via LDS tile transpose ----------------
__global__ void cvt_tr_kernel(const float* __restrict__ v, bf16_t* __restrict__ vtr) {
  __shared__ bf16_t tile[32][33];
  const int c0 = blockIdx.x * 32;
  const int s0 = blockIdx.y * 32;
  const int tx = threadIdx.x & 31;
  const int ty = threadIdx.x >> 5;      // 0..7
#pragma unroll
  for (int i = 0; i < 4; ++i) {
    int s = s0 + ty + i * 8;
    tile[ty + i * 8][tx] = f2bf(v[(size_t)s * Dd + c0 + tx]);
  }
  __syncthreads();
#pragma unroll
  for (int i = 0; i < 4; ++i) {
    int c = c0 + ty + i * 8;
    vtr[(size_t)c * Tt + s0 + tx] = tile[tx][ty + i * 8];
  }
}

// ---- main flash-attention kernel ------------------------------------------
__global__ __launch_bounds__(256, 2) void attn_fwd(
    const bf16_t* __restrict__ qb, const bf16_t* __restrict__ kb,
    const bf16_t* __restrict__ vt, float* __restrict__ out) {
  // stride-17 padding -> conflict-free column-major S slots (32 cols x 16 rows)
  __shared__ float  sred[8 * 544];   // 17408 B : per-wave partial S
  __shared__ bf16_t pbuf[16 * 40];   //  1280 B : reduced P tile (16 x 32, stride 40)
  __shared__ float  alf[16];         // per-row alpha broadcast
  __shared__ float  linv[16];        // per-row 1/l broadcast

  const int tid  = threadIdx.x;
  const int wave = tid >> 5;
  const int lane = tid & 31;
  const int half = lane >> 4;          // C/D layout: row = g + 8*half
  const int lm   = lane & 15;          // C/D layout: col = lm

  const int rt = (int)gridDim.x - 1 - (int)blockIdx.x;  // longest blocks first
  const int r0 = rt * 16;

  // Persistent Q A-fragments for this wave's 128-wide d-slice.
  // 16-bit A layout: lane<16 holds K 0..7 & 16..23, lane>=16 holds K 8..15 & 24..31.
  ABFrag qf[4];
  {
    const bf16_t* qrow = qb + (size_t)(r0 + lm) * Dd + wave * 128;
#pragma unroll
    for (int f = 0; f < 4; ++f) {
      qf[f].h[0] = *(const v8bf*)(qrow + f * 32 + half * 8);
      qf[f].h[1] = *(const v8bf*)(qrow + f * 32 + 16 + half * 8);
    }
  }

  v8f o[8];
#pragma unroll
  for (int cc = 0; cc < 8; ++cc) o[cc] = (v8f){};

  // Softmax state for the two rows this wave owns (replicated across lanes).
  const int row0g = 2 * wave;          // local row ids owned by this wave
  const int row1g = 2 * wave + 1;
  float m0 = -1e30f, m1 = -1e30f, l0 = 0.0f, l1 = 0.0f;

  const float SCALE = 0.03125f * 1.44269504088896340736f; // (1/sqrt(1024))*log2(e)

  const int ntiles = (r0 + 47) >> 5;        // 32-col KV tiles, causal bound
  float* slot = sred + wave * 544;

  for (int j = 0; j < ntiles; ++j) {
    const int s0 = j << 5;

    // ---- S phase: preload ALL 8 K B-fragments (16 loads in flight) ----
    const bf16_t* k0 = kb + (size_t)(s0 + lm) * Dd + wave * 128;        // cols s0..s0+15
    const bf16_t* k1 = kb + (size_t)(s0 + 16 + lm) * Dd + wave * 128;   // cols s0+16..s0+31
    ABFrag kf[8];
#pragma unroll
    for (int f = 0; f < 4; ++f) {
      // B layout (32x16): lane<16 holds K 0..7 & 16..23, lane>=16 holds K 8..15 & 24..31; n = lane&15
      kf[f].h[0]     = *(const v8bf*)(k0 + f * 32 + half * 8);
      kf[f].h[1]     = *(const v8bf*)(k0 + f * 32 + 16 + half * 8);
      kf[f + 4].h[0] = *(const v8bf*)(k1 + f * 32 + half * 8);
      kf[f + 4].h[1] = *(const v8bf*)(k1 + f * 32 + 16 + half * 8);
    }
    // Pin every fragment in its own registers: forces all 16 loads to be
    // issued before the WMMA chain (RA cannot coalesce into one octet, the
    // scheduler cannot sink loads below their use).
#pragma unroll
    for (int f = 0; f < 8; ++f)
      asm volatile("" : "+v"(kf[f].h[0]), "+v"(kf[f].h[1]));

    v8f sa = (v8f){}, sb = (v8f){};
#pragma unroll
    for (int f = 0; f < 4; ++f) {
      sa = __builtin_amdgcn_wmma_f32_16x16x32_bf16(false, qf[f].v, false, kf[f].v,
                                                   (short)0, sa, false, false);
      sb = __builtin_amdgcn_wmma_f32_16x16x32_bf16(false, qf[f].v, false, kf[f + 4].v,
                                                   (short)0, sb, false, false);
    }

    // ---- publish partial S (column-major, stride 17: conflict-free) ----
#pragma unroll
    for (int g = 0; g < 8; ++g) {
      slot[lm * 17 + g + 8 * half]        = sa[g];
      slot[(16 + lm) * 17 + g + 8 * half] = sb[g];
    }
    __syncthreads();

    // ---- distributed reduce + softmax: this wave owns rows 2w, 2w+1 ----
    // lane = KV column within the tile; read 2 rows from each of 8 slots.
    float x0, x1;
    {
      const float* s2 = sred + lane * 17 + row0g;
      x0 = s2[0];
      x1 = s2[1];
#pragma unroll
      for (int w2 = 1; w2 < 8; ++w2) {
        x0 += s2[w2 * 544];
        x1 += s2[w2 * 544 + 1];
      }
    }
    x0 *= SCALE;
    x1 *= SCALE;
    if (s0 + 31 > r0) {                       // causal mask (diagonal tiles only)
      const int colg = s0 + lane;
      if (colg > r0 + row0g) x0 = -1e30f;
      if (colg > r0 + row1g) x1 = -1e30f;
    }
    float mt0 = x0, mt1 = x1;
#pragma unroll
    for (int off = 1; off < 32; off <<= 1) {
      mt0 = fmaxf(mt0, __shfl_xor(mt0, off));
      mt1 = fmaxf(mt1, __shfl_xor(mt1, off));
    }
    const float mn0 = fmaxf(m0, mt0);
    const float mn1 = fmaxf(m1, mt1);
    const float al0 = exp2f(m0 - mn0);
    const float al1 = exp2f(m1 - mn1);
    const float p0 = exp2f(x0 - mn0);
    const float p1 = exp2f(x1 - mn1);
    float rs0 = p0, rs1 = p1;
#pragma unroll
    for (int off = 1; off < 32; off <<= 1) {
      rs0 += __shfl_xor(rs0, off);
      rs1 += __shfl_xor(rs1, off);
    }
    l0 = l0 * al0 + rs0;  m0 = mn0;
    l1 = l1 * al1 + rs1;  m1 = mn1;
    pbuf[row0g * 40 + lane] = f2bf(p0);
    pbuf[row1g * 40 + lane] = f2bf(p1);
    if (lane == 0) { alf[row0g] = al0; alf[row1g] = al1; }
    __syncthreads();

    // ---- everyone: fetch alpha + P A-fragment ----
    const float4 a0 = *(const float4*)(&alf[8 * half]);
    const float4 a1 = *(const float4*)(&alf[8 * half + 4]);
    const float alpha[8] = { a0.x, a0.y, a0.z, a0.w, a1.x, a1.y, a1.z, a1.w };
    ABFrag pf;
    {
      const bf16_t* pr = pbuf + lm * 40;
      pf.h[0] = *(const v8bf*)(pr + half * 8);
      pf.h[1] = *(const v8bf*)(pr + 16 + half * 8);
    }

    // ---- PV phase: preload ALL 8 V B-fragments, rescale O meanwhile ----
    ABFrag vf[8];
#pragma unroll
    for (int cc = 0; cc < 8; ++cc) {
      const bf16_t* vp = vt + (size_t)(wave * 128 + cc * 16 + lm) * Tt + s0;
      vf[cc].h[0] = *(const v8bf*)(vp + half * 8);
      vf[cc].h[1] = *(const v8bf*)(vp + 16 + half * 8);
    }
#pragma unroll
    for (int cc = 0; cc < 8; ++cc)
#pragma unroll
      for (int g = 0; g < 8; ++g) o[cc][g] *= alpha[g];
#pragma unroll
    for (int cc = 0; cc < 8; ++cc) {
      o[cc] = __builtin_amdgcn_wmma_f32_16x16x32_bf16(false, pf.v, false, vf[cc].v,
                                                      (short)0, o[cc], false, false);
    }
  }

  // ---- broadcast 1/l, normalize, write fp32 output ----
  if (lane == 0) {
    linv[row0g] = 1.0f / l0;
    linv[row1g] = 1.0f / l1;
  }
  __syncthreads();
  const float4 i0 = *(const float4*)(&linv[8 * half]);
  const float4 i1 = *(const float4*)(&linv[8 * half + 4]);
  const float inv[8] = { i0.x, i0.y, i0.z, i0.w, i1.x, i1.y, i1.z, i1.w };
#pragma unroll
  for (int cc = 0; cc < 8; ++cc) {
#pragma unroll
    for (int g = 0; g < 8; ++g) {
      out[(size_t)(r0 + g + 8 * half) * Dd + wave * 128 + cc * 16 + lm] =
          o[cc][g] * inv[g];
    }
  }
}

extern "C" void kernel_launch(void* const* d_in, const int* in_sizes, int n_in,
                              void* d_out, int out_size, void* d_ws, size_t ws_size,
                              hipStream_t stream) {
  const float* q = (const float*)d_in[0];
  const float* k = (const float*)d_in[1];
  const float* v = (const float*)d_in[2];
  float* out = (float*)d_out;

  bf16_t* qb = (bf16_t*)d_ws;                 // 16 MB
  bf16_t* kb = qb + (size_t)Tt * Dd;          // 16 MB
  bf16_t* vt = kb + (size_t)Tt * Dd;          // 16 MB (transposed V)

  const int n4blocks = (Tt * Dd) / (256 * 4); // 8192 blocks
  cvt_rm_kernel<<<n4blocks, 256, 0, stream>>>(q, qb);
  cvt_rm_kernel<<<n4blocks, 256, 0, stream>>>(k, kb);
  cvt_tr_kernel<<<dim3(Dd / 32, Tt / 32), 256, 0, stream>>>(v, vt);
  attn_fwd<<<Tt / 16, 256, 0, stream>>>(qb, kb, vt, out);
}
